// DeepFM_1391569404529
// MI455X (gfx1250) — matrix operand order repaired and checked by reading, not compiled, not run on hardware
//
#include <hip/hip_runtime.h>
#include <hip/hip_bf16.h>

// ---------------- problem constants ----------------
#define NB    16384        // batch
#define FF    39           // fields
#define V1    100001       // vocab + 1
#define EE    16           // embed dim
#define DD    624          // F*E
#define DP    640          // padded K for GEMM1 (mult of 32)
#define HH    400          // H1 == H2
#define HP    416          // padded K for GEMM2 (mult of 32)
#define EPSBN 1e-5f

// GEMM register tile: MG x NG sub-tiles of 16x16 per wave (64 x 80 output)
#define MG 4
#define NG 5

typedef __attribute__((ext_vector_type(16))) _Float16 v16h;
typedef __attribute__((ext_vector_type(8)))  float    v8f;

union Frag {
    v16h v;
    unsigned int u[8];
};

// ---------------- weight transpose + f16 convert (K-padded) ----------------
// W: K x H row-major f32  ->  Wt: H x Kp row-major f16 (zero pad K..Kp)
__global__ __launch_bounds__(256) void k_convw(const float* __restrict__ W,
                                               _Float16* __restrict__ Wt,
                                               int K, int H, int Kp) {
    int i = blockIdx.x * blockDim.x + threadIdx.x;
    if (i >= H * Kp) return;
    int h = i / Kp, k = i % Kp;
    Wt[i] = (k < K) ? (_Float16)W[(size_t)k * H + h] : (_Float16)0.0f;
}

// ---------------- gather + FM + deep-row build (one wave / sample) ----------
__global__ __launch_bounds__(256) void k_gather(const int*   __restrict__ xi,
                                                const float* __restrict__ xv,
                                                const float* __restrict__ emb1,
                                                const float* __restrict__ emb2,
                                                _Float16*    __restrict__ deep,
                                                float*       __restrict__ partial) {
    int wave = (blockIdx.x * blockDim.x + threadIdx.x) >> 5;
    int lane = threadIdx.x & 31;
    if (wave >= NB) return;
    int n = wave;

    float s[EE], sq[EE];
#pragma unroll
    for (int e = 0; e < EE; ++e) { s[e] = 0.f; sq[e] = 0.f; }
    float e1s = 0.f;

    for (int f = lane; f < FF; f += 32) {
        size_t base = (size_t)n * FF + f;
        int   idx = xi[base];
        float v   = xv[base];
        size_t row = (size_t)f * V1 + idx;
        e1s += emb1[row] * v;
        const float4* e2p = (const float4*)(emb2 + row * EE);
        _Float16* dp = deep + (size_t)n * DP + f * EE;
#pragma unroll
        for (int q = 0; q < 4; ++q) {
            float4 w4 = e2p[q];
            float w0 = w4.x * v, w1 = w4.y * v, w2 = w4.z * v, w3 = w4.w * v;
            s[4*q+0] += w0; sq[4*q+0] += w0 * w0; dp[4*q+0] = (_Float16)w0;
            s[4*q+1] += w1; sq[4*q+1] += w1 * w1; dp[4*q+1] = (_Float16)w1;
            s[4*q+2] += w2; sq[4*q+2] += w2 * w2; dp[4*q+2] = (_Float16)w2;
            s[4*q+3] += w3; sq[4*q+3] += w3 * w3; dp[4*q+3] = (_Float16)w3;
        }
    }
    // zero K-padding (columns 624..639)
    if (lane < DP - DD) deep[(size_t)n * DP + DD + lane] = (_Float16)0.0f;

    // wave32 reduction over the field dimension
#pragma unroll
    for (int off = 16; off > 0; off >>= 1) {
        e1s += __shfl_down(e1s, off, 32);
#pragma unroll
        for (int e = 0; e < EE; ++e) {
            s[e]  += __shfl_down(s[e],  off, 32);
            sq[e] += __shfl_down(sq[e], off, 32);
        }
    }
    if (lane == 0) {
        float fm = 0.f;
#pragma unroll
        for (int e = 0; e < EE; ++e) fm += 0.5f * (s[e] * s[e] - sq[e]);
        partial[n] = fm + e1s;
    }
}

// ---------------- WMMA GEMM: Z = A(NB x Kp, f16) * Bt^T + bias --------------
// Bt is H x Kp row-major f16 (pre-transposed weights).
// One wave computes a 64 x 80 output block (MG x NG tiles of 16x16):
// per K-step it loads 4 A-fragments + 5 B-fragments and issues 20 WMMAs.
__global__ __launch_bounds__(256) void k_gemm(const _Float16* __restrict__ A,
                                              const _Float16* __restrict__ Bt,
                                              const float*    __restrict__ bias,
                                              float*          __restrict__ Z,
                                              int Kp, int H) {
    int wave = (blockIdx.x * blockDim.x + threadIdx.x) >> 5;
    int lane = threadIdx.x & 31;
    int ngroups = (H >> 4) / NG;          // 5
    int gM = wave / ngroups;              // 0..NB/(16*MG)-1
    int gN = wave % ngroups;              // 0..ngroups-1
    int rc   = lane & 15;                 // A row / B column within a tile
    int half = lane >> 4;

    const unsigned int* arow[MG];
#pragma unroll
    for (int m = 0; m < MG; ++m)
        arow[m] = (const unsigned int*)(A + (size_t)((gM * MG + m) * 16 + rc) * Kp);
    const unsigned int* brow[NG];
#pragma unroll
    for (int nn = 0; nn < NG; ++nn)
        brow[nn] = (const unsigned int*)(Bt + (size_t)((gN * NG + nn) * 16 + rc) * Kp);

    v8f c[MG][NG];
#pragma unroll
    for (int m = 0; m < MG; ++m)
#pragma unroll
        for (int nn = 0; nn < NG; ++nn) c[m][nn] = (v8f){};

    for (int k0 = 0; k0 < Kp; k0 += 32) {
        int kb = k0 >> 1;                 // uint (2xf16) index of K-block
        Frag a[MG], b[NG];
#pragma unroll
        for (int i = 0; i < 8; ++i) {
            // A 16-bit 16x32 layout: VGPR i<4 -> K=2i+8*half ; i>=4 -> K=16+2(i-4)+8*half
            int ka = (i < 4 ? 2 * i : 16 + 2 * (i - 4)) + 8 * half;
#pragma unroll
            for (int m = 0; m < MG; ++m) a[m].u[i] = arow[m][kb + (ka >> 1)];
            // B 16-bit 32x16 layout (same K order as A): VGPR i<4 -> K=2i+8*half ; i>=4 -> K=16+2(i-4)+8*half, N=lane&15
            int kk = (i < 4 ? 2 * i : 16 + 2 * (i - 4)) + 8 * half;
#pragma unroll
            for (int nn = 0; nn < NG; ++nn) b[nn].u[i] = brow[nn][kb + (kk >> 1)];
        }
#pragma unroll
        for (int m = 0; m < MG; ++m)
#pragma unroll
            for (int nn = 0; nn < NG; ++nn)
                c[m][nn] = __builtin_amdgcn_wmma_f32_16x16x32_f16(
                    false, a[m].v, false, b[nn].v,
                    (short)0, c[m][nn], false, false);
    }

#pragma unroll
    for (int nn = 0; nn < NG; ++nn) {
        int col = (gN * NG + nn) * 16 + rc;
        float bv = bias[col];
#pragma unroll
        for (int m = 0; m < MG; ++m) {
#pragma unroll
            for (int r = 0; r < 8; ++r) {
                // C/D f32 layout: M = r + 8*half, N = lane&15
                Z[(size_t)((gM * MG + m) * 16 + r + 8 * half) * H + col] = c[m][nn][r] + bv;
            }
        }
    }
}

// ---------------- batch-norm column statistics (coalesced) ------------------
// One block per 16 columns; lanes stride columns for coalesced loads.
__global__ __launch_bounds__(256) void k_bnstats(const float* __restrict__ Z,
                                                 float* __restrict__ mu,
                                                 float* __restrict__ rs,
                                                 int H) {
    __shared__ float ssum[256];
    __shared__ float ssq[256];
    int t  = threadIdx.x;
    int jc = t & 15;          // column within the 16-column group
    int rr = t >> 4;          // row slice 0..15
    int j  = blockIdx.x * 16 + jc;
    float a = 0.f, b = 0.f;
    for (int r = rr; r < NB; r += 16) {
        float v = Z[(size_t)r * H + j];
        a += v; b += v * v;
    }
    ssum[t] = a; ssq[t] = b;
    __syncthreads();
    for (int sdist = 128; sdist >= 16; sdist >>= 1) {
        if (t < sdist) { ssum[t] += ssum[t + sdist]; ssq[t] += ssq[t + sdist]; }
        __syncthreads();
    }
    if (t < 16) {
        float m   = ssum[t] * (1.0f / NB);
        float var = ssq[t] * (1.0f / NB) - m * m;
        mu[j] = m;
        rs[j] = rsqrtf(var + EPSBN);
    }
}

// ---------------- BN apply + f16 convert (K-padded output) ------------------
__global__ __launch_bounds__(256) void k_bnapply(const float* __restrict__ Z,
                                                 const float* __restrict__ g,
                                                 const float* __restrict__ be,
                                                 const float* __restrict__ mu,
                                                 const float* __restrict__ rs,
                                                 _Float16* __restrict__ Hout,
                                                 int H, int Kp) {
    long long i = (long long)blockIdx.x * blockDim.x + threadIdx.x;
    if (i >= (long long)NB * Kp) return;
    int r = (int)(i / Kp), j = (int)(i % Kp);
    _Float16 o = (_Float16)0.0f;
    if (j < H)
        o = (_Float16)(g[j] * (Z[(size_t)r * H + j] - mu[j]) * rs[j] + be[j]);
    Hout[i] = o;
}

// ---------------- fused BN2 + row-sum + partial + bias ----------------------
__global__ __launch_bounds__(256) void k_final(const float* __restrict__ partial,
                                               const float* __restrict__ Z2,
                                               const float* __restrict__ g,
                                               const float* __restrict__ be,
                                               const float* __restrict__ mu,
                                               const float* __restrict__ rs,
                                               const float* __restrict__ bias,
                                               float* __restrict__ out) {
    int wave = (blockIdx.x * blockDim.x + threadIdx.x) >> 5;
    int lane = threadIdx.x & 31;
    if (wave >= NB) return;
    float acc = 0.f;
    for (int j = lane; j < HH; j += 32) {
        float z = Z2[(size_t)wave * HH + j];
        acc += g[j] * (z - mu[j]) * rs[j] + be[j];
    }
#pragma unroll
    for (int off = 16; off > 0; off >>= 1) acc += __shfl_down(acc, off, 32);
    if (lane == 0) out[wave] = partial[wave] + acc + bias[0];
}

// ---------------- host launcher ---------------------------------------------
extern "C" void kernel_launch(void* const* d_in, const int* in_sizes, int n_in,
                              void* d_out, int out_size, void* d_ws, size_t ws_size,
                              hipStream_t stream) {
    const int*   xi   = (const int*)  d_in[0];
    const float* xv   = (const float*)d_in[1];
    const float* emb1 = (const float*)d_in[2];
    const float* emb2 = (const float*)d_in[3];
    const float* W1   = (const float*)d_in[4];
    const float* b1   = (const float*)d_in[5];
    const float* g1   = (const float*)d_in[6];
    const float* be1  = (const float*)d_in[7];
    const float* W2   = (const float*)d_in[8];
    const float* b2   = (const float*)d_in[9];
    const float* g2   = (const float*)d_in[10];
    const float* be2  = (const float*)d_in[11];
    const float* bias = (const float*)d_in[12];
    float* out = (float*)d_out;

    // workspace carve-out (256B aligned chunks)
    char* ws = (char*)d_ws;
    size_t off = 0;
    auto carve = [&](size_t bytes) {
        char* p = ws + off;
        off += (bytes + 255) & ~(size_t)255;
        return p;
    };
    _Float16* deepA = (_Float16*)carve((size_t)NB * DP * sizeof(_Float16)); // 21 MB
    _Float16* W1t   = (_Float16*)carve((size_t)HH * DP * sizeof(_Float16)); // 0.5 MB
    _Float16* W2t   = (_Float16*)carve((size_t)HH * HP * sizeof(_Float16)); // 0.33 MB
    float*    Z1    = (float*)   carve((size_t)NB * HH * sizeof(float));    // 26 MB
    _Float16* h1    = (_Float16*)carve((size_t)NB * HP * sizeof(_Float16)); // 14 MB
    float*    Z2    = (float*)   carve((size_t)NB * HH * sizeof(float));    // 26 MB
    float*    part  = (float*)   carve((size_t)NB * sizeof(float));
    float*    mu1   = (float*)   carve(HH * sizeof(float));
    float*    rs1   = (float*)   carve(HH * sizeof(float));
    float*    mu2   = (float*)   carve(HH * sizeof(float));
    float*    rs2   = (float*)   carve(HH * sizeof(float));

    // 1) weight convert + transpose (f16, K-padded)
    k_convw<<<(HH * DP + 255) / 256, 256, 0, stream>>>(W1, W1t, DD, HH, DP);
    k_convw<<<(HH * HP + 255) / 256, 256, 0, stream>>>(W2, W2t, HH, HH, HP);

    // 2) gather + FM + deep rows : one wave per sample (8 waves / block)
    k_gather<<<NB / 8, 256, 0, stream>>>(xi, xv, emb1, emb2, deepA, part);

    // 3) GEMM1: waves = (NB/16/MG) * (HH/16/NG) = 256*5 = 1280 -> 160 blocks
    int gemm_blocks = ((NB / 16 / MG) * ((HH / 16) / NG)) / 8;
    k_gemm<<<gemm_blocks, 256, 0, stream>>>(deepA, W1t, b1, Z1, DP, HH);

    // 4) BN1 stats + apply (-> f16, K-padded to 416)
    k_bnstats<<<HH / 16, 256, 0, stream>>>(Z1, mu1, rs1, HH);
    k_bnapply<<<(int)(((long long)NB * HP + 255) / 256), 256, 0, stream>>>(
        Z1, g1, be1, mu1, rs1, h1, HH, HP);

    // 5) GEMM2: (NB x HP) x (HP x HH) -> Z2
    k_gemm<<<gemm_blocks, 256, 0, stream>>>(h1, W2t, b2, Z2, HP, HH);

    // 6) BN2 stats + fused final reduction
    k_bnstats<<<HH / 16, 256, 0, stream>>>(Z2, mu2, rs2, HH);
    k_final<<<NB / 8, 256, 0, stream>>>(part, Z2, g2, be2, mu2, rs2, bias, out);
}
